// LRMUCell_19155554140581
// MI455X (gfx1250) — matrix operand order repaired and checked by reading, not compiled, not run on hardware
//
#include <hip/hip_runtime.h>

// ---------------------------------------------------------------------------
// LMU scan for MI455X (gfx1250, wave32, WMMA).
//   ys[b,t,:] = scan of  u = x_t@W_in + m@W_mem + bias ;  m = m@Ad + u (x) Bd
// Ad/Bd are computed on-device via expm (scaling & squaring + Taylor).
// Main kernel: 8 persistent WGs x 512 threads; state m held in LDS as
// pre-split bf16 hi/lo; recurrence uses bf16x3 (hi*hi + lo*hi + hi*lo)
// v_wmma_f32_16x16x32_bf16 for fp32-class accuracy at bf16 throughput.
// ---------------------------------------------------------------------------

#define MD      32
#define OD      128
#define DIM     128
#define TSTEPS  512
#define BATCH   64
#define NB      8                    // batches per workgroup
#define NWG     (BATCH / NB)         // 8 workgroups
#define NTHREADS 512                 // 16 waves (wave32)

#define M_ROWS      (NB * MD)        // 256 state rows per WG
#define M_ROWS_PAD  (M_ROWS + 1)     // +1 permanent zero row (GEMM2 row pad)
#define M_STRIDE    136              // bf16 halves per row (128 + 8 pad)
#define AD_STRIDE   136              // bf16 halves per AdT row

typedef __attribute__((ext_vector_type(16))) __bf16 v16bf;
typedef __attribute__((ext_vector_type(8)))  __bf16 v8bf;
typedef __attribute__((ext_vector_type(8)))  float  v8f;
typedef __attribute__((ext_vector_type(4)))  float  v4f;

#define WMMA_BF16(A, B, C) \
  __builtin_amdgcn_wmma_f32_16x16x32_bf16(false, (A), false, (B), (short)0, (C), false, false)

// ---- LDS layout (bytes) for main kernel -----------------------------------
#define OFF_MHI   0
#define SZ_M      (M_ROWS_PAD * M_STRIDE * 2)          // 257*136*2 = 69904
#define OFF_MLO   (OFF_MHI + SZ_M)                     // 69904
#define OFF_ADH   (OFF_MLO + SZ_M)                     // 139808
#define SZ_AD     (OD * AD_STRIDE * 2)                 // 34816
#define OFF_ADL   (OFF_ADH + SZ_AD)                    // 174624
#define OFF_RED   (OFF_ADL + SZ_AD)                    // 209440
#define SZ_RED    (2 * 8 * 16 * 16 * 4)                // 16384
#define OFF_UROW  (OFF_RED + SZ_RED)                   // 225824
#define OFF_BD    (OFF_UROW + 256 * 4)                 // 226848
#define OFF_BIAS  (OFF_BD + OD * 4)                    // 227360
#define SMEM_MAIN (OFF_BIAS + MD * 4)                  // 227488 B  (< 320 KB)

union FragBF { v16bf v; v8bf h[2]; };

// ===========================================================================
// Setup kernel 1: compute Ad = expm(em)[:128,:128], Bd = expm(em)[128,:128]
// em is the 129x129 LMU generator; scaling 2^-8, order-12 Taylor (Horner),
// then 8 squarings. Single workgroup, all in LDS.
// ===========================================================================
#define EN 129
#define ES 130
#define SMEM_EXPM (3 * EN * ES * 4)   // 201240 B

__global__ __launch_bounds__(1024) void lmu_expm_kernel(float* __restrict__ ws_Ad,
                                                        float* __restrict__ ws_Bd) {
  extern __shared__ char smem[];
  float* S = (float*)smem;          // scaled generator
  float* P = S + EN * ES;           // accumulator
  float* T = P + EN * ES;           // matmul temp
  const int tid = threadIdx.x;
  const int nth = blockDim.x;

  // Build em (em[r][c] = A[c][r]/theta for r<128; row 128 = Bt), scale by 2^-8.
  for (int idx = tid; idx < EN * EN; idx += nth) {
    int r = idx / EN, c = idx % EN;
    float v = 0.0f;
    if (c < 128) {
      float R = 2.0f * (float)c + 1.0f;
      if (r < 128) {
        float sgn = (c < r) ? -1.0f : ((((c - r + 1) & 1) != 0) ? -1.0f : 1.0f);
        v = sgn * R * (1.0f / 512.0f);
      } else {
        v = (((c & 1) != 0) ? -1.0f : 1.0f) * R * (1.0f / 512.0f);
      }
    }
    S[r * ES + c] = v * (1.0f / 256.0f);
    P[r * ES + c] = (r == c) ? 1.0f : 0.0f;
  }
  __syncthreads();

  // Horner Taylor: for k = 12..1 : P = I + (S @ P) / k
  for (int k = 12; k >= 1; --k) {
    for (int idx = tid; idx < EN * EN; idx += nth) {
      int i2 = idx / EN, j2 = idx % EN;
      float acc = 0.0f;
      for (int kx = 0; kx < EN; ++kx) acc += S[i2 * ES + kx] * P[kx * ES + j2];
      T[i2 * ES + j2] = acc;
    }
    __syncthreads();
    float invk = 1.0f / (float)k;
    for (int idx = tid; idx < EN * EN; idx += nth) {
      int i2 = idx / EN, j2 = idx % EN;
      P[i2 * ES + j2] = ((i2 == j2) ? 1.0f : 0.0f) + T[i2 * ES + j2] * invk;
    }
    __syncthreads();
  }
  // Square 8 times.
  for (int it = 0; it < 8; ++it) {
    for (int idx = tid; idx < EN * EN; idx += nth) {
      int i2 = idx / EN, j2 = idx % EN;
      float acc = 0.0f;
      for (int kx = 0; kx < EN; ++kx) acc += P[i2 * ES + kx] * P[kx * ES + j2];
      T[i2 * ES + j2] = acc;
    }
    __syncthreads();
    for (int idx = tid; idx < EN * EN; idx += nth) {
      int i2 = idx / EN, j2 = idx % EN;
      P[i2 * ES + j2] = T[i2 * ES + j2];
    }
    __syncthreads();
  }
  for (int idx = tid; idx < 128 * 128; idx += nth) {
    int r = idx >> 7, c = idx & 127;
    ws_Ad[idx] = P[r * ES + c];
  }
  for (int c = tid; c < 128; c += nth) ws_Bd[c] = P[128 * ES + c];
}

// ===========================================================================
// Setup kernel 2: transpose + convert W_mem (4096x32) -> WmT bf16 (32x4096)
// and W_in (128x32) -> WinT bf16 (32x128): B-fragment friendly (col-major).
// ===========================================================================
__global__ void lmu_pack_kernel(const float* __restrict__ W_in,
                                const float* __restrict__ W_mem,
                                __bf16* __restrict__ WmT,
                                __bf16* __restrict__ WinT) {
  int idx = blockIdx.x * blockDim.x + threadIdx.x;
  if (idx < 32 * 4096) {
    int n = idx >> 12, k = idx & 4095;
    WmT[idx] = (__bf16)W_mem[k * 32 + n];
  }
  int idx2 = idx - 32 * 4096;
  if (idx2 >= 0 && idx2 < 32 * 128) {
    int n = idx2 >> 7, k = idx2 & 127;
    WinT[idx2] = (__bf16)W_in[k * 32 + n];
  }
}

// ===========================================================================
// Main persistent scan kernel.
// ===========================================================================
__global__ __launch_bounds__(NTHREADS) void lrmu_scan_kernel(
    const float*  __restrict__ x,     // (64, 512, 128)
    const float*  __restrict__ bias,  // (32,)
    const float*  __restrict__ Ad,    // (128,128) fp32 from expm
    const float*  __restrict__ Bd,    // (128,)
    const __bf16* __restrict__ WmT,   // (32, 4096) bf16
    const __bf16* __restrict__ WinT,  // (32, 128) bf16
    float* __restrict__ out)          // (64, 512, 4096)
{
  extern __shared__ char smem[];
  __bf16* mhi  = (__bf16*)(smem + OFF_MHI);
  __bf16* mlo  = (__bf16*)(smem + OFF_MLO);
  __bf16* adh  = (__bf16*)(smem + OFF_ADH);
  __bf16* adl  = (__bf16*)(smem + OFF_ADL);
  float*  red  = (float*)(smem + OFF_RED);
  float*  urow = (float*)(smem + OFF_UROW);
  float*  bdv  = (float*)(smem + OFF_BD);
  float*  bis  = (float*)(smem + OFF_BIAS);

  const int tid  = threadIdx.x;
  const int wave = tid >> 5;
  const int lane = tid & 31;
  const int l15  = lane & 15;
  const int hi16 = lane >> 4;      // 0 | 1
  const int kbA  = hi16 * 8;       // A-frag K base (16-bit A layout)
  const int kbB  = hi16 * 16;      // B-frag K base
  const int wgb  = blockIdx.x * NB;

  // ---- init: zero state (incl. permanent zero row), split AdT hi/lo ----
  for (int i = tid; i < M_ROWS_PAD * M_STRIDE; i += NTHREADS) {
    mhi[i] = (__bf16)0.0f; mlo[i] = (__bf16)0.0f;
  }
  for (int i = tid; i < OD * OD; i += NTHREADS) {
    int k = i >> 7, j = i & 127;
    float a = Ad[i];                       // Ad[k][j]
    __bf16 h = (__bf16)a;
    __bf16 l = (__bf16)(a - (float)h);
    adh[j * AD_STRIDE + k] = h;            // col-major for B-fragments
    adl[j * AD_STRIDE + k] = l;
  }
  for (int i = tid; i < OD; i += NTHREADS) bdv[i] = Bd[i];
  for (int i = tid; i < MD; i += NTHREADS) bis[i] = bias[i];
  __syncthreads();

  const int nt2 = wave & 1;   // GEMM2: output n-tile of this wave
  const int kc  = wave >> 1;  // GEMM2: k-chunk of this wave (16 ksteps)

  for (int t = 0; t < TSTEPS; ++t) {
    // ============ Phase A: partial U = m@W_mem (+ x@W_in on waves 0,1) ====
    v8f c2 = {0.f, 0.f, 0.f, 0.f, 0.f, 0.f, 0.f, 0.f};
    {
      const int bb = l15;                                   // batch row of A
      const __bf16* wb = WmT + (size_t)(nt2 * 16 + l15) * 4096;
#pragma unroll 4
      for (int s = 0; s < 16; ++s) {
        int kk = kc * 16 + s;            // flat k-step (32 wide), 0..127
        int i  = kk >> 2;                // which md row of m
        int j0 = (kk & 3) * 32;          // col offset within row
        int row = (bb < NB) ? (bb * MD + i) : M_ROWS;       // pad -> zero row
        const __bf16* mrh = mhi + row * M_STRIDE + j0 + kbA;
        const __bf16* mrl = mlo + row * M_STRIDE + j0 + kbA;
        FragBF ah, al, ub;
        ah.h[0] = *(const v8bf*)(mrh);        ah.h[1] = *(const v8bf*)(mrh + 16);
        al.h[0] = *(const v8bf*)(mrl);        al.h[1] = *(const v8bf*)(mrl + 16);
        ub.h[0] = *(const v8bf*)(wb + kk * 32 + kbA);
        ub.h[1] = *(const v8bf*)(wb + kk * 32 + kbA + 16);
        __builtin_prefetch(wb + (kk + 1) * 32, 0, 3);
        c2 = WMMA_BF16(ah.v, ub.v, c2);       // m_hi @ W
        c2 = WMMA_BF16(al.v, ub.v, c2);       // m_lo @ W
      }
    }
    if (wave < 2) {                           // x_t @ W_in folded in
      const int  bb  = l15;
      const bool val = (bb < NB);
      const int  bg  = wgb + (val ? bb : 0);
      const float*  xr = x + ((size_t)bg * TSTEPS + t) * DIM;
      const __bf16* wi = WinT + (size_t)(nt2 * 16 + l15) * DIM;
#pragma unroll
      for (int kk = 0; kk < 4; ++kk) {
        int c0 = kk * 32 + kbA;
        v4f x0 = *(const v4f*)(xr + c0);
        v4f x1 = *(const v4f*)(xr + c0 + 4);
        v4f x2 = *(const v4f*)(xr + c0 + 16);
        v4f x3 = *(const v4f*)(xr + c0 + 20);
        v16bf ax;
#pragma unroll
        for (int e = 0; e < 4; ++e) {
          ax[e]      = (__bf16)(val ? x0[e] : 0.0f);
          ax[4 + e]  = (__bf16)(val ? x1[e] : 0.0f);
          ax[8 + e]  = (__bf16)(val ? x2[e] : 0.0f);
          ax[12 + e] = (__bf16)(val ? x3[e] : 0.0f);
        }
        FragBF ub;
        ub.h[0] = *(const v8bf*)(wi + kk * 32 + kbA);
        ub.h[1] = *(const v8bf*)(wi + kk * 32 + kbA + 16);
        c2 = WMMA_BF16(ax, ub.v, c2);
      }
    }
    {   // stash partial C-fragment: red[nt][kc][M][n]
      float* rp = red + (((nt2 * 8 + kc) * 16 + hi16 * 8) * 16 + l15);
#pragma unroll
      for (int v = 0; v < 8; ++v) rp[v * 16] = c2[v];
    }
    __syncthreads();

    // ============ Phase B: reduce partials -> u_row[b*32 + k] =============
    if (tid < 256) {
      int b = tid >> 5, kap = tid & 31;
      int nt = kap >> 4, nl = kap & 15;
      float s = bis[kap];
#pragma unroll
      for (int k2 = 0; k2 < 8; ++k2) s += red[((nt * 8 + k2) * 16 + b) * 16 + nl];
      urow[b * MD + kap] = s;
    }
    __syncthreads();

    // ============ Phase C: newM = m@Ad (bf16x3) + u (x) Bd ================
    {
      const int rbase = wave * 16;            // this wave's 16 state rows
      const int row   = rbase + l15;
      const __bf16* mrh = mhi + row * M_STRIDE;
      const __bf16* mrl = mlo + row * M_STRIDE;
      v16bf ah[4], al[4];                     // preload A frags (own rows only)
#pragma unroll
      for (int kk = 0; kk < 4; ++kk) {
        int c0 = kk * 32 + kbA;
        FragBF uh, ul;
        uh.h[0] = *(const v8bf*)(mrh + c0);   uh.h[1] = *(const v8bf*)(mrh + c0 + 16);
        ul.h[0] = *(const v8bf*)(mrl + c0);   ul.h[1] = *(const v8bf*)(mrl + c0 + 16);
        ah[kk] = uh.v; al[kk] = ul.v;
      }
      float uv[8];
      const int r0 = rbase + hi16 * 8;
#pragma unroll
      for (int v = 0; v < 8; ++v) uv[v] = urow[r0 + v];

#pragma unroll 2
      for (int nt = 0; nt < 8; ++nt) {
        const int j = nt * 16 + l15;
        const float bdj = bdv[j];
        v8f c;
#pragma unroll
        for (int v = 0; v < 8; ++v) c[v] = uv[v] * bdj;   // init with u (x) Bd
        const __bf16* bh = adh + (size_t)j * AD_STRIDE;
        const __bf16* bl = adl + (size_t)j * AD_STRIDE;
#pragma unroll
        for (int kk = 0; kk < 4; ++kk) {
          FragBF uB, uL;
          uB.h[0] = *(const v8bf*)(bh + kk * 32 + kbA);
          uB.h[1] = *(const v8bf*)(bh + kk * 32 + kbA + 16);
          uL.h[0] = *(const v8bf*)(bl + kk * 32 + kbA);
          uL.h[1] = *(const v8bf*)(bl + kk * 32 + kbA + 16);
          c = WMMA_BF16(ah[kk], uB.v, c);     // hi*hi
          c = WMMA_BF16(al[kk], uB.v, c);     // lo*hi
          c = WMMA_BF16(ah[kk], uL.v, c);     // hi*lo  (bf16x3 ~ fp32)
        }
        // writeback: fp32 -> global out; hi/lo split -> LDS for next step
#pragma unroll
        for (int v = 0; v < 8; ++v) {
          int r = r0 + v;
          float valf = c[v];
          __bf16 h = (__bf16)valf;
          __bf16 l = (__bf16)(valf - (float)h);
          mhi[r * M_STRIDE + j] = h;
          mlo[r * M_STRIDE + j] = l;
          int b = r >> 5, ii = r & 31;
          out[((size_t)(wgb + b) * TSTEPS + t) * (MD * OD) + (size_t)ii * OD + j] = valf;
        }
      }
    }
    __syncthreads();
  }
}

// ===========================================================================
extern "C" void kernel_launch(void* const* d_in, const int* in_sizes, int n_in,
                              void* d_out, int out_size, void* d_ws, size_t ws_size,
                              hipStream_t stream) {
  (void)in_sizes; (void)n_in; (void)out_size; (void)ws_size;
  const float* x     = (const float*)d_in[0];   // (64,512,128)
  const float* W_in  = (const float*)d_in[1];   // (128,32)
  const float* W_mem = (const float*)d_in[2];   // (4096,32)
  const float* bias  = (const float*)d_in[3];   // (32,)
  float* out = (float*)d_out;

  char* ws = (char*)d_ws;
  float*  ws_Ad   = (float*)ws;                     // 128*128 f32
  float*  ws_Bd   = ws_Ad + 128 * 128;              // 128 f32
  __bf16* ws_WmT  = (__bf16*)(ws_Bd + 128);         // 32*4096 bf16
  __bf16* ws_WinT = ws_WmT + 32 * 4096;             // 32*128 bf16
  // total ws: 66048 + 262144 + 8192 = 336384 bytes

  lmu_expm_kernel<<<1, 1024, SMEM_EXPM, stream>>>(ws_Ad, ws_Bd);
  lmu_pack_kernel<<<(32 * 4096 + 32 * 128 + 255) / 256, 256, 0, stream>>>(
      W_in, W_mem, ws_WmT, ws_WinT);
  lrmu_scan_kernel<<<NWG, NTHREADS, SMEM_MAIN, stream>>>(
      x, bias, ws_Ad, ws_Bd, ws_WmT, ws_WinT, out);
}
